// a_RTF_ISTA_model_62483184222572
// MI455X (gfx1250) — matrix operand order repaired and checked by reading, not compiled, not run on hardware
//
#include <hip/hip_runtime.h>

// ---------------------------------------------------------------------------
// ISTA recurrence on MI455X (gfx1250):
//   Wy = y @ W^T ; d0 = 0
//   repeat 16x:  z = d @ S^T + Wy ; u = z @ D^T ; d = soft_thr(u, thr) @ D
// GEMMs run as split-bf16 (hi+lo) WMMA with f32 accumulation:
//   A*B ~= Ah*Bh + Ah*Bl + Al*Bh   (3x v_wmma_f32_16x16x32_bf16 per tile)
// B panels are staged into LDS by the Tensor Data Mover (tensor_load_to_lds,
// TENSORcnt double-buffering, padded rows to dodge LDS bank conflicts);
// A activations stream from global (L2-resident).
// ---------------------------------------------------------------------------

typedef __attribute__((ext_vector_type(16))) __bf16        v16bf;
typedef __attribute__((ext_vector_type(8)))  float         v8f;
typedef __attribute__((ext_vector_type(4)))  unsigned int  u32x4;
typedef __attribute__((ext_vector_type(4)))  int           i32x4;
typedef __attribute__((ext_vector_type(8)))  int           i32x8;
typedef unsigned short u16;

static constexpr int Bn   = 8192;  // batch rows
static constexpr int Mdim = 256;   // y feature dim
static constexpr int Ndim = 512;   // state dim
static constexpr int ITER = 16;

// TDM staging geometry: chunk = 128 cols x 128 halves, padded rows in LDS
static constexpr int KC      = 128;               // halves per chunk
static constexpr int ROW_PAD = 32;                // bytes of pad per row
static constexpr int ROW_B   = KC * 2 + ROW_PAD;  // 288 B LDS row stride
static constexpr int PANEL_B = 128 * ROW_B;       // 36864 B per panel buffer

#if __has_builtin(__builtin_amdgcn_tensor_load_to_lds)
#define HAVE_TDM 1
#else
#define HAVE_TDM 0
#endif

union Frag { v16bf v; u32x4 u[2]; };

__device__ __forceinline__ u16 f32_to_bf16_rne(float x) {
  unsigned b = __float_as_uint(x);
  unsigned r = (b + 0x7fffu + ((b >> 16) & 1u)) >> 16;
  return (u16)r;
}
__device__ __forceinline__ float bf16_bits_to_f32(u16 h) {
  return __uint_as_float(((unsigned)h) << 16);
}

// ---------------------------------------------------------------------------
// split f32 -> (hi, lo) bf16 pair, elementwise
// ---------------------------------------------------------------------------
__global__ void split_bf16_kernel(const float* __restrict__ in,
                                  u16* __restrict__ hi, u16* __restrict__ lo,
                                  int n) {
  int i = blockIdx.x * blockDim.x + threadIdx.x;
  if (i >= n) return;
  float x  = in[i];
  u16   h  = f32_to_bf16_rne(x);
  hi[i] = h;
  lo[i] = f32_to_bf16_rne(x - bf16_bits_to_f32(h));
}

// Dt[n][k] = D[k][n], split into hi/lo (512 x 512)
__global__ void transpose_split_kernel(const float* __restrict__ D,
                                       u16* __restrict__ th, u16* __restrict__ tl) {
  int i = blockIdx.x * blockDim.x + threadIdx.x;      // i = n*512 + k
  if (i >= Ndim * Ndim) return;
  int n = i >> 9, k = i & 511;
  float x  = D[k * Ndim + n];
  u16   h  = f32_to_bf16_rne(x);
  th[i] = h;
  tl[i] = f32_to_bf16_rne(x - bf16_bits_to_f32(h));
}

#if HAVE_TDM
// ---------------------------------------------------------------------------
// TDM: DMA a 2D tile (rows x cols bf16, global row stride = rstride halves)
// into LDS at lds_off, inserting ROW_PAD bytes after each 256 B row.
// D# layout per CDNA5 ISA ch.8 (group0 128b, group1 256b; groups 2/3 unused
// for 2D tiles). Issued wave-uniform from wave 0 only (TDM ignores EXEC).
// ---------------------------------------------------------------------------
__device__ __forceinline__ void tdm_load_panel(const u16* gsrc, unsigned lds_off,
                                               int rows, int cols, int rstride) {
  unsigned long long ga = (unsigned long long)gsrc;
  u32x4 g0;
  g0[0] = 1u;                                     // count=1, user descriptor
  g0[1] = lds_off;                                // lds_addr (bytes)
  g0[2] = (unsigned)(ga & 0xffffffffull);         // global_addr[31:0]
  g0[3] = (unsigned)((ga >> 32) & 0x01ffffffull)  // global_addr[56:32]
          | (2u << 30);                           // type = 2 ("image")
  const unsigned pad_interval = 5u;               // 64 DWORDs = 256 B (one row)
  const unsigned pad_amount   = 7u;               // 8 DWORDs  = 32 B
  i32x8 g1;
  g1[0] = (int)((1u << 16)                        // data_size = 2 B
                | (1u << 20)                      // pad_enable
                | (pad_interval << 22) | (pad_amount << 25));
  g1[1] = (int)(((unsigned)cols & 0xffffu) << 16);            // tensor_dim0 lo
  g1[2] = (int)((((unsigned)cols >> 16) & 0xffffu)            // tensor_dim0 hi
                | (((unsigned)rows & 0xffffu) << 16));        // tensor_dim1 lo
  g1[3] = (int)((((unsigned)rows >> 16) & 0xffffu)            // tensor_dim1 hi
                | (((unsigned)cols & 0xffffu) << 16));        // tile_dim0
  g1[4] = (int)((unsigned)rows & 0xffffu);                    // tile_dim1 (dim2=0)
  g1[5] = (int)rstride;                                       // dim0_stride lo32
  g1[6] = 0;
  g1[7] = 0;
  i32x4 z4 = {0, 0, 0, 0};
#if defined(__clang_major__) && __clang_major__ >= 23
  i32x8 z8 = {0, 0, 0, 0, 0, 0, 0, 0};
  __builtin_amdgcn_tensor_load_to_lds(g0, g1, z4, z4, z8, 0);
#else
  __builtin_amdgcn_tensor_load_to_lds(g0, g1, z4, z4, 0);
#endif
}
#endif  // HAVE_TDM

// ---------------------------------------------------------------------------
// Split-bf16 WMMA GEMM:  C[M x N] = A[M x K] * Bmat, Bmat stored k-contiguous
// per output column: Bmat[n*K + k].  Epilogue: optional +bias, soft-threshold,
// f32 store, hi/lo re-split.  Block tile 128x128, 8 waves; wave = 2x4 tiles.
// ---------------------------------------------------------------------------
__global__ __launch_bounds__(256)
void gemm_bf16x3_kernel(const u16* __restrict__ Ah, const u16* __restrict__ Al,
                        const u16* __restrict__ Bh, const u16* __restrict__ Bl,
                        const float* __restrict__ bias,   // nullable
                        const float* __restrict__ thrp,   // nullable -> soft_thr
                        float* __restrict__ outF,         // nullable
                        u16* __restrict__ outH, u16* __restrict__ outL, // nullable
                        int M, int N, int K) {
  const int lane = threadIdx.x & 31;
  const int wid  = threadIdx.x >> 5;
  const int wm   = wid & 3;           // 4 wave-rows
  const int wn   = wid >> 2;          // 2 wave-cols
  const int m0   = blockIdx.y * 128 + wm * 32;
  const int n0b  = blockIdx.x * 128;  // block panel base column
  const int lrow = lane & 15;
  const bool hiH = lane >= 16;
  const int kbA  = hiH ? 8 : 0;       // A frag K-sub-base (ISA 16-bit A layout)
  const int kbB  = hiH ? 8 : 0;       // B frag K-sub-base (same layout as A)

  v8f acc[2][4] = {};

#if HAVE_TDM
  __shared__ __align__(16) unsigned char smem[4 * PANEL_B];  // [buf][hi/lo]
  const unsigned smbase = (unsigned)(unsigned long long)&smem[0];
  const int nchunk = K / KC;
  if (wid == 0) {
    tdm_load_panel(Bh + (size_t)n0b * K, smbase + 0 * PANEL_B, 128, KC, K);
    tdm_load_panel(Bl + (size_t)n0b * K, smbase + 1 * PANEL_B, 128, KC, K);
  }
  for (int c = 0; c < nchunk; ++c) {
    const int cb = c * KC;
    if (wid == 0) __builtin_amdgcn_s_wait_tensorcnt(0);
    __syncthreads();                           // chunk c visible to all waves
    if (wid == 0 && c + 1 < nchunk) {          // prefetch chunk c+1
      const unsigned nb = ((c + 1) & 1) ? 2u : 0u;
      tdm_load_panel(Bh + (size_t)n0b * K + cb + KC, smbase + (nb + 0) * PANEL_B, 128, KC, K);
      tdm_load_panel(Bl + (size_t)n0b * K + cb + KC, smbase + (nb + 1) * PANEL_B, 128, KC, K);
    }
    const unsigned char* ph = smem + ((c & 1) ? 2 : 0) * PANEL_B;
    const unsigned char* pl = ph + PANEL_B;
#pragma unroll
    for (int k0 = 0; k0 < KC; k0 += 32) {
      Frag ah[2], al[2], bh4[4], bl4[4];
#pragma unroll
      for (int i = 0; i < 2; ++i) {            // A from global (streaming)
        size_t ro = (size_t)(m0 + i * 16 + lrow) * K + (cb + k0 + kbA);
        ah[i].u[0] = *reinterpret_cast<const u32x4*>(Ah + ro);
        ah[i].u[1] = *reinterpret_cast<const u32x4*>(Ah + ro + 16);
        al[i].u[0] = *reinterpret_cast<const u32x4*>(Al + ro);
        al[i].u[1] = *reinterpret_cast<const u32x4*>(Al + ro + 16);
      }
#pragma unroll
      for (int j = 0; j < 4; ++j) {            // B from padded LDS panel
        const unsigned off = (unsigned)(wn * 64 + j * 16 + lrow) * ROW_B + (k0 + kbB) * 2;
        bh4[j].u[0] = *reinterpret_cast<const u32x4*>(ph + off);
        bh4[j].u[1] = *reinterpret_cast<const u32x4*>(ph + off + 32);
        bl4[j].u[0] = *reinterpret_cast<const u32x4*>(pl + off);
        bl4[j].u[1] = *reinterpret_cast<const u32x4*>(pl + off + 32);
      }
#pragma unroll
      for (int i = 0; i < 2; ++i)
#pragma unroll
        for (int j = 0; j < 4; ++j) {
          acc[i][j] = __builtin_amdgcn_wmma_f32_16x16x32_bf16(
              false, ah[i].v, false, bh4[j].v, (short)0, acc[i][j], false, false);
          acc[i][j] = __builtin_amdgcn_wmma_f32_16x16x32_bf16(
              false, ah[i].v, false, bl4[j].v, (short)0, acc[i][j], false, false);
          acc[i][j] = __builtin_amdgcn_wmma_f32_16x16x32_bf16(
              false, al[i].v, false, bh4[j].v, (short)0, acc[i][j], false, false);
        }
    }
    __syncthreads();                           // done reading buf (c&1)
  }
#else  // fallback: B fragments direct from global
  for (int k0 = 0; k0 < K; k0 += 32) {
    Frag ah[2], al[2], bh4[4], bl4[4];
#pragma unroll
    for (int i = 0; i < 2; ++i) {
      size_t ro = (size_t)(m0 + i * 16 + lrow) * K + (k0 + kbA);
      ah[i].u[0] = *reinterpret_cast<const u32x4*>(Ah + ro);
      ah[i].u[1] = *reinterpret_cast<const u32x4*>(Ah + ro + 16);
      al[i].u[0] = *reinterpret_cast<const u32x4*>(Al + ro);
      al[i].u[1] = *reinterpret_cast<const u32x4*>(Al + ro + 16);
    }
#pragma unroll
    for (int j = 0; j < 4; ++j) {
      size_t co = (size_t)(n0b + wn * 64 + j * 16 + lrow) * K + (k0 + kbB);
      bh4[j].u[0] = *reinterpret_cast<const u32x4*>(Bh + co);
      bh4[j].u[1] = *reinterpret_cast<const u32x4*>(Bh + co + 16);
      bl4[j].u[0] = *reinterpret_cast<const u32x4*>(Bl + co);
      bl4[j].u[1] = *reinterpret_cast<const u32x4*>(Bl + co + 16);
    }
#pragma unroll
    for (int i = 0; i < 2; ++i)
#pragma unroll
      for (int j = 0; j < 4; ++j) {
        acc[i][j] = __builtin_amdgcn_wmma_f32_16x16x32_bf16(
            false, ah[i].v, false, bh4[j].v, (short)0, acc[i][j], false, false);
        acc[i][j] = __builtin_amdgcn_wmma_f32_16x16x32_bf16(
            false, ah[i].v, false, bl4[j].v, (short)0, acc[i][j], false, false);
        acc[i][j] = __builtin_amdgcn_wmma_f32_16x16x32_bf16(
            false, al[i].v, false, bh4[j].v, (short)0, acc[i][j], false, false);
      }
  }
#endif

  // -------- epilogue --------
  const float thr  = thrp ? thrp[0] : 0.0f;
  const int   moff = hiH ? 8 : 0;
#pragma unroll
  for (int i = 0; i < 2; ++i)
#pragma unroll
    for (int j = 0; j < 4; ++j) {
      const int nc = n0b + wn * 64 + j * 16 + lrow;
#pragma unroll
      for (int v = 0; v < 8; ++v) {
        const int    r   = m0 + i * 16 + moff + v;
        const size_t idx = (size_t)r * N + nc;
        float x = acc[i][j][v];
        if (bias) x += bias[idx];
        if (thrp) x = fmaxf(x - thr, 0.0f) - fmaxf(-x - thr, 0.0f);
        if (outF) outF[idx] = x;
        if (outH) {
          u16 h = f32_to_bf16_rne(x);
          outH[idx] = h;
          outL[idx] = f32_to_bf16_rne(x - bf16_bits_to_f32(h));
        }
      }
    }
}

// ---------------------------------------------------------------------------
// Host-side orchestration
// ---------------------------------------------------------------------------
extern "C" void kernel_launch(void* const* d_in, const int* in_sizes, int n_in,
                              void* d_out, int out_size, void* d_ws, size_t ws_size,
                              hipStream_t stream) {
  const float* y   = (const float*)d_in[0];   // (8192, 256)
  const float* S   = (const float*)d_in[1];   // (512, 512)
  const float* W   = (const float*)d_in[2];   // (512, 256)
  const float* D   = (const float*)d_in[3];   // (512, 512)
  const float* thr = (const float*)d_in[4];   // (1,)
  float*       out = (float*)d_out;           // (17, 8192, 512)

  char* ws = (char*)d_ws;
  size_t off = 0;
  auto alloc = [&](size_t bytes) -> char* {
    char* p = ws + off;
    off += (bytes + 255) & ~(size_t)255;
    return p;
  };
  const size_t actE = (size_t)Bn * Ndim;
  u16* yh  = (u16*)alloc((size_t)Bn * Mdim * 2);
  u16* yl  = (u16*)alloc((size_t)Bn * Mdim * 2);
  u16* Wh  = (u16*)alloc((size_t)Ndim * Mdim * 2);
  u16* Wl  = (u16*)alloc((size_t)Ndim * Mdim * 2);
  u16* Sh  = (u16*)alloc((size_t)Ndim * Ndim * 2);
  u16* Sl  = (u16*)alloc((size_t)Ndim * Ndim * 2);
  u16* Dh  = (u16*)alloc((size_t)Ndim * Ndim * 2);
  u16* Dl  = (u16*)alloc((size_t)Ndim * Ndim * 2);
  u16* Dth = (u16*)alloc((size_t)Ndim * Ndim * 2);
  u16* Dtl = (u16*)alloc((size_t)Ndim * Ndim * 2);
  float* Wy = (float*)alloc(actE * 4);
  u16* dh = (u16*)alloc(actE * 2);
  u16* dl = (u16*)alloc(actE * 2);
  u16* zh = (u16*)alloc(actE * 2);
  u16* zl = (u16*)alloc(actE * 2);
  u16* sh = (u16*)alloc(actE * 2);
  u16* sl = (u16*)alloc(actE * 2);
  if (off > ws_size) return;  // deterministic bail-out if scratch too small

  auto ceil256 = [](int n) { return (n + 255) / 256; };
  split_bf16_kernel<<<ceil256(Bn * Mdim), 256, 0, stream>>>(y, yh, yl, Bn * Mdim);
  split_bf16_kernel<<<ceil256(Ndim * Mdim), 256, 0, stream>>>(W, Wh, Wl, Ndim * Mdim);
  split_bf16_kernel<<<ceil256(Ndim * Ndim), 256, 0, stream>>>(S, Sh, Sl, Ndim * Ndim);
  split_bf16_kernel<<<ceil256(Ndim * Ndim), 256, 0, stream>>>(D, Dh, Dl, Ndim * Ndim);
  transpose_split_kernel<<<ceil256(Ndim * Ndim), 256, 0, stream>>>(D, Dth, Dtl);

  const dim3 grid(Ndim / 128, Bn / 128);  // (4, 64)

  // Wy = y @ W^T (K = 256)
  gemm_bf16x3_kernel<<<grid, 256, 0, stream>>>(
      yh, yl, Wh, Wl, nullptr, nullptr, Wy, nullptr, nullptr, Bn, Ndim, Mdim);

  // d0 = 0
  hipMemsetAsync(out, 0, actE * sizeof(float), stream);
  hipMemsetAsync(dh, 0, actE * sizeof(u16), stream);
  hipMemsetAsync(dl, 0, actE * sizeof(u16), stream);

  for (int it = 0; it < ITER; ++it) {
    // z = d @ S^T + Wy
    gemm_bf16x3_kernel<<<grid, 256, 0, stream>>>(
        dh, dl, Sh, Sl, Wy, nullptr, nullptr, zh, zl, Bn, Ndim, Ndim);
    // s = soft_thr(z @ D^T, thr)
    gemm_bf16x3_kernel<<<grid, 256, 0, stream>>>(
        zh, zl, Dh, Dl, nullptr, thr, nullptr, sh, sl, Bn, Ndim, Ndim);
    // d = s @ D  -> f32 output slice it+1 and bf16 state
    gemm_bf16x3_kernel<<<grid, 256, 0, stream>>>(
        sh, sl, Dth, Dtl, nullptr, nullptr,
        out + (size_t)(it + 1) * actE, dh, dl, Bn, Ndim, Ndim);
  }
}
